// RnnGruModel_27573690040799
// MI455X (gfx1250) — compile-verified, matrix operand order checked by reading, not run on hardware
//
#include <hip/hip_runtime.h>
#include <hip/hip_bf16.h>

typedef __attribute__((ext_vector_type(16))) _Float16 v16h;
typedef __attribute__((ext_vector_type(8)))  float    v8f;
typedef __attribute__((ext_vector_type(4)))  float    v4f;
typedef __attribute__((ext_vector_type(4)))  _Float16 v4h;

#define T_DIM 256
#define B_DIM 128
#define F_DIM 1600
#define M_DIM (T_DIM * B_DIM)   // 32768 rows
#define H1_DIM 60
#define NPAD 64                 // H1 padded to 64 cols
#define MT 32                   // rows per block
#define BK 64                   // K chunk staged in LDS

// ---------------------------------------------------------------------------
// Kernel 1: h1 = tanh(X @ W1^T + b1), f16 WMMA, fp32 accumulate.
// Block = 256 threads = 8 waves; wave (mt,nt) owns one 16x16 output tile of a
// 32-row x 64-col block tile. W1 pad rows 60..63 are zeroed once up front.
// Staging uses b128 global loads + b64 LDS stores (4x f16) — the kernel is
// HBM-bound (210 MB of X at 23.3 TB/s ~ 9 us floor), so loads must be wide
// and the staging loop must stay branch-free.
// ---------------------------------------------------------------------------
__global__ __launch_bounds__(256)
void mlp1_wmma(const float* __restrict__ X, const float* __restrict__ W1,
               const float* __restrict__ b1, float* __restrict__ H1out) {
    __shared__ _Float16 Xs[MT][BK];     // 4 KB
    __shared__ _Float16 Ws[NPAD][BK];   // 8 KB

    const int tid  = threadIdx.x;
    const int lane = tid & 31;
    const int wv   = tid >> 5;      // 0..7
    const int mt   = wv >> 2;       // 0..1  (16-row tile within block)
    const int nt   = wv & 3;        // 0..3  (16-col tile)
    const int half = lane >> 4;     // K-subblock select per ISA layout
    const int rl   = lane & 15;     // row (A) / col (B) index within tile
    const long rowBase = (long)blockIdx.x * MT;

    // Zero W pad rows 60..63 once (4*64 = 256 halves, 1 per thread).
    // Visibility is covered by the first __syncthreads() in the K-loop.
    Ws[60 + (tid >> 6)][tid & 63] = (_Float16)0.0f;

    v8f acc = {};

    for (int kk = 0; kk < F_DIM; kk += BK) {
        // Stage X chunk (32x64 f32 -> f16): 512 float4, 2 per thread.
        #pragma unroll
        for (int u = 0; u < 2; ++u) {
            int idx = tid + u * 256;
            int r = idx >> 4, c4 = idx & 15;
            v4f x4 = *reinterpret_cast<const v4f*>(
                &X[(rowBase + r) * F_DIM + kk + c4 * 4]);
            v4h h4;
            h4[0] = (_Float16)x4[0]; h4[1] = (_Float16)x4[1];
            h4[2] = (_Float16)x4[2]; h4[3] = (_Float16)x4[3];
            *reinterpret_cast<v4h*>(&Xs[r][c4 * 4]) = h4;
        }
        // Stage W1 chunk (60 rows x 64 cols = 960 float4): threads 0..239
        // each own 4 contiguous float4 of one row. Single guard, divergent
        // only in wave 7; all loads inside are unconditional.
        if (tid < 240) {
            int r  = tid >> 2;            // 0..59
            int c0 = (tid & 3) * 4;       // float4 column base: 0,4,8,12
            #pragma unroll
            for (int u = 0; u < 4; ++u) {
                v4f w4 = *reinterpret_cast<const v4f*>(
                    &W1[r * F_DIM + kk + (c0 + u) * 4]);
                v4h h4;
                h4[0] = (_Float16)w4[0]; h4[1] = (_Float16)w4[1];
                h4[2] = (_Float16)w4[2]; h4[3] = (_Float16)w4[3];
                *reinterpret_cast<v4h*>(&Ws[r][(c0 + u) * 4]) = h4;
            }
        }
        if (kk + BK < F_DIM) {  // pull next X chunk toward the caches
            __builtin_prefetch(
                &X[(rowBase + (tid >> 4)) * F_DIM + kk + BK + (tid & 15) * 4],
                0, 1);
        }
        __syncthreads();

        #pragma unroll
        for (int ks = 0; ks < BK; ks += 32) {
            v16h a, b;
            #pragma unroll
            for (int e = 0; e < 16; ++e) {
                // 16-bit A 16x32 lane/VGPR layout (ISA 7.12.2):
                // VGPR v<4 holds K = 2v..2v+1 (+8 for lanes 16..31),
                // VGPR v>=4 holds K = 16+2(v-4).. (+8 for lanes 16..31).
                int v = e >> 1, lo = e & 1;
                int k = ((v < 4) ? (2 * v + 8 * half)
                                 : (16 + 2 * (v - 4) + 8 * half)) + lo;
                a[e] = Xs[mt * 16 + rl][ks + k];
                b[e] = Ws[nt * 16 + rl][ks + k];  // B col-major mirrors A layout
            }
            acc = __builtin_amdgcn_wmma_f32_16x16x32_f16(
                false, a, false, b, (short)0, acc, false, false);
        }
        __syncthreads();
    }

    // C/D layout: lane n = nt*16 + rl ; VGPR r -> row m = mt*16 + half*8 + r.
    const int n = nt * 16 + rl;
    const float bias = (n < H1_DIM) ? b1[n] : 0.0f;
    #pragma unroll
    for (int r = 0; r < 8; ++r) {
        long m = rowBase + mt * 16 + half * 8 + r;
        H1out[m * NPAD + n] = tanhf(acc[r] + bias);
    }
}

// ---------------------------------------------------------------------------
// Kernel 2: per-row 60 -> tanh(15) -> 10 -> GRU-L0 input gates (15), [M][16].
// ---------------------------------------------------------------------------
__global__ __launch_bounds__(256)
void mlp_tail(const float* __restrict__ H1,
              const float* __restrict__ W2, const float* __restrict__ b2,
              const float* __restrict__ W3, const float* __restrict__ b3,
              const float* __restrict__ w_ih0, const float* __restrict__ b_ih0,
              float* __restrict__ GI0) {
    long m = (long)blockIdx.x * blockDim.x + threadIdx.x;
    if (m >= M_DIM) return;

    float h[H1_DIM];
    #pragma unroll
    for (int q = 0; q < H1_DIM / 4; ++q) {          // 15x b128 loads
        v4f v = *reinterpret_cast<const v4f*>(&H1[m * NPAD + q * 4]);
        h[q * 4 + 0] = v[0]; h[q * 4 + 1] = v[1];
        h[q * 4 + 2] = v[2]; h[q * 4 + 3] = v[3];
    }

    float a2[15];
    #pragma unroll
    for (int i = 0; i < 15; ++i) {
        float s = b2[i];
        #pragma unroll 4
        for (int j = 0; j < H1_DIM; ++j) s += W2[i * H1_DIM + j] * h[j];
        a2[i] = tanhf(s);
    }
    float a3[10];
    #pragma unroll
    for (int i = 0; i < 10; ++i) {
        float s = b3[i];
        #pragma unroll
        for (int j = 0; j < 15; ++j) s += W3[i * 15 + j] * a2[j];
        a3[i] = s;  // no activation on last MLP layer
    }
    float gi[16];
    #pragma unroll
    for (int g = 0; g < 15; ++g) {
        float s = b_ih0[g];
        #pragma unroll
        for (int j = 0; j < 10; ++j) s += w_ih0[g * 10 + j] * a3[j];
        gi[g] = s;
    }
    gi[15] = 0.0f;
    #pragma unroll
    for (int q = 0; q < 4; ++q) {                   // 4x b128 stores
        v4f v = {gi[q * 4], gi[q * 4 + 1], gi[q * 4 + 2], gi[q * 4 + 3]};
        *reinterpret_cast<v4f*>(&GI0[m * 16 + q * 4]) = v;
    }
}

// ---------------------------------------------------------------------------
// Kernel 3: 3-layer GRU scan (H=5) + output projection. One lane per batch.
// ---------------------------------------------------------------------------
__device__ __forceinline__ float sigm_(float x) { return 1.0f / (1.0f + __expf(-x)); }

__device__ __forceinline__ void gru_step(float h[5], const float gi[15],
                                         const float* __restrict__ whh,
                                         const float* __restrict__ bhh) {
    float gh[15];
    #pragma unroll
    for (int g = 0; g < 15; ++g) {
        float s = bhh[g];
        #pragma unroll
        for (int j = 0; j < 5; ++j) s += whh[g * 5 + j] * h[j];
        gh[g] = s;
    }
    float hn[5];
    #pragma unroll
    for (int i = 0; i < 5; ++i) {
        float r = sigm_(gi[i] + gh[i]);
        float z = sigm_(gi[5 + i] + gh[5 + i]);
        float n = tanhf(gi[10 + i] + r * gh[10 + i]);
        hn[i] = (1.0f - z) * n + z * h[i];
    }
    #pragma unroll
    for (int i = 0; i < 5; ++i) h[i] = hn[i];
}

__global__ __launch_bounds__(128)
void gru_scan(const float* __restrict__ GI0,
              const float* __restrict__ w_hh0, const float* __restrict__ b_hh0,
              const float* __restrict__ w_ih1, const float* __restrict__ w_hh1,
              const float* __restrict__ b_ih1, const float* __restrict__ b_hh1,
              const float* __restrict__ w_ih2, const float* __restrict__ w_hh2,
              const float* __restrict__ b_ih2, const float* __restrict__ b_hh2,
              const float* __restrict__ Wout, const float* __restrict__ bout,
              float* __restrict__ Y) {
    __shared__ float swhh[3][75], sbhh[3][15];
    __shared__ float swih[2][75], sbih[2][15];
    __shared__ float swout[5], sbout;

    const int tid = threadIdx.x;
    const float* whhp[3] = {w_hh0, w_hh1, w_hh2};
    const float* bhhp[3] = {b_hh0, b_hh1, b_hh2};
    const float* wihp[2] = {w_ih1, w_ih2};
    const float* bihp[2] = {b_ih1, b_ih2};
    for (int l = 0; l < 3; ++l) {
        for (int i = tid; i < 75; i += 128) swhh[l][i] = whhp[l][i];
        for (int i = tid; i < 15; i += 128) sbhh[l][i] = bhhp[l][i];
    }
    for (int l = 0; l < 2; ++l) {
        for (int i = tid; i < 75; i += 128) swih[l][i] = wihp[l][i];
        for (int i = tid; i < 15; i += 128) sbih[l][i] = bihp[l][i];
    }
    if (tid < 5) swout[tid] = Wout[tid];
    if (tid == 0) sbout = bout[0];
    __syncthreads();

    const int b = tid;  // batch element
    float h0[5] = {}, h1v[5] = {}, h2v[5] = {};

    for (int t = 0; t < T_DIM; ++t) {
        // layer 0: input gates precomputed; 4x b128 loads
        float gi[16];
        const float* gp = &GI0[((long)t * B_DIM + b) * 16];
        #pragma unroll
        for (int q = 0; q < 4; ++q) {
            v4f v = *reinterpret_cast<const v4f*>(&gp[q * 4]);
            gi[q * 4 + 0] = v[0]; gi[q * 4 + 1] = v[1];
            gi[q * 4 + 2] = v[2]; gi[q * 4 + 3] = v[3];
        }
        gru_step(h0, gi, swhh[0], sbhh[0]);

        // layer 1
        float gi1[15];
        #pragma unroll
        for (int g = 0; g < 15; ++g) {
            float s = sbih[0][g];
            #pragma unroll
            for (int j = 0; j < 5; ++j) s += swih[0][g * 5 + j] * h0[j];
            gi1[g] = s;
        }
        gru_step(h1v, gi1, swhh[1], sbhh[1]);

        // layer 2
        float gi2[15];
        #pragma unroll
        for (int g = 0; g < 15; ++g) {
            float s = sbih[1][g];
            #pragma unroll
            for (int j = 0; j < 5; ++j) s += swih[1][g * 5 + j] * h1v[j];
            gi2[g] = s;
        }
        gru_step(h2v, gi2, swhh[2], sbhh[2]);

        float y = sbout;
        #pragma unroll
        for (int j = 0; j < 5; ++j) y += swout[j] * h2v[j];
        Y[(long)t * B_DIM + b] = y;   // output layout (T, 1, B)
    }
}

// ---------------------------------------------------------------------------
extern "C" void kernel_launch(void* const* d_in, const int* in_sizes, int n_in,
                              void* d_out, int out_size, void* d_ws, size_t ws_size,
                              hipStream_t stream) {
    const float* X     = (const float*)d_in[0];
    const float* W1    = (const float*)d_in[1];
    const float* b1    = (const float*)d_in[2];
    const float* W2    = (const float*)d_in[3];
    const float* b2    = (const float*)d_in[4];
    const float* W3    = (const float*)d_in[5];
    const float* b3    = (const float*)d_in[6];
    const float* w_ih0 = (const float*)d_in[7];
    const float* w_hh0 = (const float*)d_in[8];
    const float* b_ih0 = (const float*)d_in[9];
    const float* b_hh0 = (const float*)d_in[10];
    const float* w_ih1 = (const float*)d_in[11];
    const float* w_hh1 = (const float*)d_in[12];
    const float* b_ih1 = (const float*)d_in[13];
    const float* b_hh1 = (const float*)d_in[14];
    const float* w_ih2 = (const float*)d_in[15];
    const float* w_hh2 = (const float*)d_in[16];
    const float* b_ih2 = (const float*)d_in[17];
    const float* b_hh2 = (const float*)d_in[18];
    const float* Wout  = (const float*)d_in[19];
    const float* bout  = (const float*)d_in[20];
    (void)in_sizes; (void)n_in; (void)out_size; (void)ws_size;

    float* H1  = (float*)d_ws;                                     // [M][64] = 8 MB
    float* GI0 = (float*)((char*)d_ws + (size_t)M_DIM * NPAD * 4); // [M][16] = 2 MB
    float* Y   = (float*)d_out;                                    // [T][B]

    mlp1_wmma<<<M_DIM / MT, 256, 0, stream>>>(X, W1, b1, H1);
    mlp_tail<<<M_DIM / 256, 256, 0, stream>>>(H1, W2, b2, W3, b3, w_ih0, b_ih0, GI0);
    gru_scan<<<1, 128, 0, stream>>>(GI0,
                                    w_hh0, b_hh0,
                                    w_ih1, w_hh1, b_ih1, b_hh1,
                                    w_ih2, w_hh2, b_ih2, b_hh2,
                                    Wout, bout, Y);
}
